// MambaBlock_44702019617596
// MI455X (gfx1250) — matrix operand order repaired and checked by reading, not compiled, not run on hardware
//
#include <hip/hip_runtime.h>
#include <hip/hip_bf16.h>
#include <cstdint>

// ---- problem constants (match reference) ----
#define B_      4
#define L_      1024
#define DMODEL  768
#define DSTATE  16
#define DCONV   4
#define DINNER  1536          // 2 * 768
#define DTRANK  48
#define DTPAD   64            // dt_rank padded up so K is a multiple of 32 for WMMA
#define NROWS   (B_ * L_)     // 4096 token rows
#define EPSV    1e-5f
#define XDBL_W  (DTRANK + 2 * DSTATE)   // 80

typedef __attribute__((ext_vector_type(16))) __bf16 v16bf;
typedef __attribute__((ext_vector_type(8)))  __bf16 bf16x8;
typedef __attribute__((ext_vector_type(8)))  float  v8f;

// ---------------------------------------------------------------------------
// fp32 -> bf16 cast with optional column slice + zero padding.
// ---------------------------------------------------------------------------
__global__ void cast_pad_bf16_kernel(const float* __restrict__ src, __bf16* __restrict__ dst,
                                     int rows, int ksrc, int src_ld, int src_col0, int kdst) {
    int idx = blockIdx.x * blockDim.x + threadIdx.x;
    if (idx >= rows * kdst) return;
    int r = idx / kdst, c = idx % kdst;
    float v = (c < ksrc) ? src[(size_t)r * src_ld + src_col0 + c] : 0.0f;
    dst[idx] = (__bf16)v;
}

// ---------------------------------------------------------------------------
// RMSNorm over d_model, emitting bf16 rows. One 256-thread block per token.
// ---------------------------------------------------------------------------
__global__ void rmsnorm_bf16_kernel(const float* __restrict__ x, const float* __restrict__ w,
                                    __bf16* __restrict__ out) {
    __shared__ float red[256];
    const int row = blockIdx.x;
    const float* xr = x + (size_t)row * DMODEL;
    float s = 0.f;
    for (int c = threadIdx.x; c < DMODEL; c += 256) { float v = xr[c]; s += v * v; }
    red[threadIdx.x] = s;
    __syncthreads();
    for (int off = 128; off > 0; off >>= 1) {
        if (threadIdx.x < off) red[threadIdx.x] += red[threadIdx.x + off];
        __syncthreads();
    }
    const float scale = rsqrtf(red[0] / (float)DMODEL + EPSV);
    __bf16* orow = out + (size_t)row * DMODEL;
    for (int c = threadIdx.x; c < DMODEL; c += 256)
        orow[c] = (__bf16)(xr[c] * scale * w[c]);
}

// ---------------------------------------------------------------------------
// Register-blocked WMMA GEMM: C[M,N] = A[M,K] * Bt[N,K]^T (+ optional fp32
// residual), bf16 inputs, f32 accumulate (v_wmma_f32_16x16x32_bf16).
// Each wave computes a (WM*16) x (WN*16) tile: WM*WN independent accumulator
// chains, WM+WN fragment loads per K-step -> high WMMA:VMEM ratio and ILP on
// the matrix pipe. Fragment gathers follow ISA 7.12.2 (wave32):
//   A : lane<16 -> row M=lane, K {0..7,16..23}; lane>=16 -> K {8..15,24..31}
//   B : lane holds column N=lane&15, K same as A: lane<16 {0..7,16..23}; lane>=16 {8..15,24..31}
//   D : element i -> row i + 8*(lane>>4), col lane&15
// Requires M % (16*WM) == 0, N % (16*WN) == 0, K % 32 == 0.
// ---------------------------------------------------------------------------
template<int WM, int WN>
__global__ __launch_bounds__(256)
void wmma_gemm_bf16_kernel(const __bf16* __restrict__ A, const __bf16* __restrict__ Bt,
                           const float* __restrict__ resid, float* __restrict__ C,
                           int M, int N, int K) {
    const int lane   = threadIdx.x & 31;
    const int wave   = threadIdx.x >> 5;
    const int tilesN = N / (16 * WN);
    const int tilesT = (M / (16 * WM)) * tilesN;
    const int tile   = blockIdx.x * 8 + wave;
    if (tile >= tilesT) return;          // wave-uniform: WMMA never runs with partial EXEC
    const int tm = (tile / tilesN) * (16 * WM);
    const int tn = (tile % tilesN) * (16 * WN);
    const int r  = lane & 15;
    const int hi = lane >> 4;

    const __bf16* aptr[WM];
    const __bf16* bptr[WN];
#pragma unroll
    for (int m = 0; m < WM; ++m) aptr[m] = A  + (size_t)(tm + m * 16 + r) * K + hi * 8;
#pragma unroll
    for (int n = 0; n < WN; ++n) bptr[n] = Bt + (size_t)(tn + n * 16 + r) * K + hi * 8;

    v8f acc[WM][WN];
#pragma unroll
    for (int m = 0; m < WM; ++m)
#pragma unroll
        for (int n = 0; n < WN; ++n) acc[m][n] = (v8f){};

    for (int k0 = 0; k0 < K; k0 += 32) {
        if (k0 + 128 < K) {              // gfx1250 global_prefetch hints, 256B ahead
            __builtin_prefetch(aptr[0] + k0 + 128, 0, 1);
            __builtin_prefetch(bptr[0] + k0 + 128, 0, 1);
        }
        v16bf av[WM], bv[WN];
#pragma unroll
        for (int m = 0; m < WM; ++m) {
            bf16x8 a0 = *(const bf16x8*)(aptr[m] + k0);
            bf16x8 a1 = *(const bf16x8*)(aptr[m] + k0 + 16);
            av[m] = __builtin_shufflevector(a0, a1, 0,1,2,3,4,5,6,7,8,9,10,11,12,13,14,15);
        }
#pragma unroll
        for (int n = 0; n < WN; ++n) {
            bf16x8 b0 = *(const bf16x8*)(bptr[n] + k0);
            bf16x8 b1 = *(const bf16x8*)(bptr[n] + k0 + 16);
            bv[n] = __builtin_shufflevector(b0, b1, 0,1,2,3,4,5,6,7,8,9,10,11,12,13,14,15);
        }
#pragma unroll
        for (int m = 0; m < WM; ++m)
#pragma unroll
            for (int n = 0; n < WN; ++n)
                acc[m][n] = __builtin_amdgcn_wmma_f32_16x16x32_bf16(
                    false, av[m], false, bv[n], (short)0, acc[m][n], false, false);
    }

#pragma unroll
    for (int m = 0; m < WM; ++m) {
#pragma unroll
        for (int n = 0; n < WN; ++n) {
            const int col = tn + n * 16 + r;
#pragma unroll
            for (int i = 0; i < 8; ++i) {
                size_t idx = (size_t)(tm + m * 16 + i + 8 * hi) * N + col;
                float v = acc[m][n][i];
                if (resid) v += resid[idx];
                C[idx] = v;
            }
        }
    }
}

// ---------------------------------------------------------------------------
// Causal depthwise conv1d (k=4) + SiLU on the x half of xz.
// ---------------------------------------------------------------------------
__global__ void conv_silu_kernel(const float* __restrict__ xz, const float* __restrict__ cw,
                                 const float* __restrict__ cb,
                                 float* __restrict__ xf, __bf16* __restrict__ xh) {
    int idx = blockIdx.x * blockDim.x + threadIdx.x;
    if (idx >= NROWS * DINNER) return;
    const int d   = idx % DINNER;
    const int row = idx / DINNER;          // b*L + t
    const int t   = row % L_;
    float s = cb[d];
#pragma unroll
    for (int i = 0; i < DCONV; ++i) {
        int ti = t - (DCONV - 1) + i;
        if (ti >= 0)
            s += xz[(size_t)(row - t + ti) * (2 * DINNER) + d] * cw[d * DCONV + i];
    }
    float act = s / (1.f + __expf(-s));    // SiLU
    xf[idx] = act;
    xh[idx] = (__bf16)act;
}

// ---------------------------------------------------------------------------
// Fused selective scan + D skip + SiLU(z) gate. One lane per (b,d) channel,
// 16-wide state in registers; per-timestep B/C vectors broadcast through LDS.
// ---------------------------------------------------------------------------
__global__ void scan_kernel(const float* __restrict__ dt, const float* __restrict__ xf,
                            const float* __restrict__ xdbl, const float* __restrict__ xz,
                            const float* __restrict__ A_log, const float* __restrict__ Dp,
                            __bf16* __restrict__ y) {
    __shared__ float Bs[DSTATE], Cs[DSTATE];
    const int b = blockIdx.x / (DINNER / 256);
    const int d = (blockIdx.x % (DINNER / 256)) * 256 + threadIdx.x;

    float Arow[DSTATE], h[DSTATE];
#pragma unroll
    for (int n = 0; n < DSTATE; ++n) {
        Arow[n] = -__expf(A_log[d * DSTATE + n]);   // A = -exp(A_log)
        h[n]    = 0.f;
    }
    const float dval = Dp[d];

    for (int t = 0; t < L_; ++t) {
        const size_t row = (size_t)b * L_ + t;
        if (threadIdx.x < 2 * DSTATE) {   // lanes 0..15 -> B, 16..31 -> C (cols 48..79)
            float v = xdbl[row * XDBL_W + DTRANK + threadIdx.x];
            if (threadIdx.x < DSTATE) Bs[threadIdx.x] = v;
            else                      Cs[threadIdx.x - DSTATE] = v;
        }
        __syncthreads();

        const float dtv = dt[row * DINNER + d];
        const float xv  = xf[row * DINNER + d];
        const float zv  = xz[row * (2 * DINNER) + DINNER + d];
        const float sp  = (dtv > 20.f) ? dtv : log1pf(__expf(dtv));   // softplus
        const float du  = dtv * xv;                                    // raw dt in deltaB_u
        float yv = 0.f;
#pragma unroll
        for (int n = 0; n < DSTATE; ++n) {
            float dA = __expf(sp * Arow[n]);
            h[n] = dA * h[n] + du * Bs[n];
            yv  += h[n] * Cs[n];
        }
        yv = (yv + xv * dval) * (zv / (1.f + __expf(-zv)));   // D skip + SiLU gate
        y[row * DINNER + d] = (__bf16)yv;
        __syncthreads();
    }
}

// ---------------------------------------------------------------------------
static inline size_t align256(size_t x) { return (x + 255) & ~(size_t)255; }

extern "C" void kernel_launch(void* const* d_in, const int* in_sizes, int n_in,
                              void* d_out, int out_size, void* d_ws, size_t ws_size,
                              hipStream_t stream) {
    (void)in_sizes; (void)n_in; (void)out_size; (void)ws_size;

    const float* hidden    = (const float*)d_in[0];   // (4,1024,768)
    const float* norm_w    = (const float*)d_in[1];   // (768)
    const float* in_proj_w = (const float*)d_in[2];   // (3072,768)
    const float* conv_w    = (const float*)d_in[3];   // (1536,4)
    const float* conv_b    = (const float*)d_in[4];   // (1536)
    const float* x_proj_w  = (const float*)d_in[5];   // (80,1536)
    const float* dt_proj_w = (const float*)d_in[6];   // (1536,48)
    const float* A_log     = (const float*)d_in[7];   // (1536,16)
    const float* D_param   = (const float*)d_in[8];   // (1536)
    const float* out_proj_w= (const float*)d_in[9];   // (768,1536)
    float* out = (float*)d_out;                       // (4,1024,768)

    // ---- workspace carve-up (~142 MB; fits comfortably in the 192MB L2) ----
    char* ws = (char*)d_ws;
    size_t off = 0;
    auto carve = [&](size_t bytes) { char* p = ws + off; off += align256(bytes); return p; };
    __bf16* inw_bf   = (__bf16*)carve((size_t)2 * DINNER * DMODEL * 2); // 3072x768
    __bf16* xpw_bf   = (__bf16*)carve((size_t)XDBL_W * DINNER * 2);     // 80x1536
    __bf16* dtw_bf   = (__bf16*)carve((size_t)DINNER * DTPAD * 2);      // 1536x64 (zero-pad K)
    __bf16* outw_bf  = (__bf16*)carve((size_t)DMODEL * DINNER * 2);     // 768x1536
    __bf16* h_bf     = (__bf16*)carve((size_t)NROWS * DMODEL * 2);      // normed tokens
    float*  xz       = (float*)carve((size_t)NROWS * 2 * DINNER * 4);   // in_proj output
    float*  x_f32    = (float*)carve((size_t)NROWS * DINNER * 4);       // conv+SiLU fp32
    __bf16* x_bf     = (__bf16*)carve((size_t)NROWS * DINNER * 2);      // conv+SiLU bf16
    float*  xdbl     = (float*)carve((size_t)NROWS * XDBL_W * 4);       // x_proj output
    __bf16* dtlow_bf = (__bf16*)carve((size_t)NROWS * DTPAD * 2);       // dt slice, padded
    float*  dt_full  = (float*)carve((size_t)NROWS * DINNER * 4);       // dt_proj output
    __bf16* y_bf     = (__bf16*)carve((size_t)NROWS * DINNER * 2);      // gated scan output

    const int TPB = 256;
    auto grid1d = [](size_t n, int tpb) { return (unsigned)((n + tpb - 1) / tpb); };

    // 1) weights -> bf16 (dt_proj zero-padded from K=48 to K=64)
    cast_pad_bf16_kernel<<<grid1d((size_t)2*DINNER*DMODEL, TPB), TPB, 0, stream>>>(
        in_proj_w, inw_bf, 2*DINNER, DMODEL, DMODEL, 0, DMODEL);
    cast_pad_bf16_kernel<<<grid1d((size_t)XDBL_W*DINNER, TPB), TPB, 0, stream>>>(
        x_proj_w, xpw_bf, XDBL_W, DINNER, DINNER, 0, DINNER);
    cast_pad_bf16_kernel<<<grid1d((size_t)DINNER*DTPAD, TPB), TPB, 0, stream>>>(
        dt_proj_w, dtw_bf, DINNER, DTRANK, DTRANK, 0, DTPAD);
    cast_pad_bf16_kernel<<<grid1d((size_t)DMODEL*DINNER, TPB), TPB, 0, stream>>>(
        out_proj_w, outw_bf, DMODEL, DINNER, DINNER, 0, DINNER);

    // 2) RMSNorm -> bf16
    rmsnorm_bf16_kernel<<<NROWS, TPB, 0, stream>>>(hidden, norm_w, h_bf);

    // 3) xz = h @ in_proj_w^T        (4096x768x3072), 64x64 wave tiles
    {
        int tiles = (NROWS/64) * ((2*DINNER)/64);
        wmma_gemm_bf16_kernel<4,4><<<(tiles + 7)/8, TPB, 0, stream>>>(
            h_bf, inw_bf, nullptr, xz, NROWS, 2*DINNER, DMODEL);
    }

    // 4) causal depthwise conv + SiLU
    conv_silu_kernel<<<grid1d((size_t)NROWS*DINNER, TPB), TPB, 0, stream>>>(
        xz, conv_w, conv_b, x_f32, x_bf);

    // 5) x_dbl = x @ x_proj_w^T      (4096x1536x80), 32x80 wave tiles
    {
        int tiles = (NROWS/32) * (XDBL_W/80);
        wmma_gemm_bf16_kernel<2,5><<<(tiles + 7)/8, TPB, 0, stream>>>(
            x_bf, xpw_bf, nullptr, xdbl, NROWS, XDBL_W, DINNER);
    }

    // 6) slice dt columns [0,48) -> bf16, zero-pad to 64
    cast_pad_bf16_kernel<<<grid1d((size_t)NROWS*DTPAD, TPB), TPB, 0, stream>>>(
        xdbl, dtlow_bf, NROWS, DTRANK, XDBL_W, 0, DTPAD);

    // 7) dt = dt_low @ dt_proj_w^T   (4096x64x1536; pad contributes zeros)
    {
        int tiles = (NROWS/64) * (DINNER/64);
        wmma_gemm_bf16_kernel<4,4><<<(tiles + 7)/8, TPB, 0, stream>>>(
            dtlow_bf, dtw_bf, nullptr, dt_full, NROWS, DINNER, DTPAD);
    }

    // 8) fused selective scan + D skip + SiLU(z) gate -> bf16
    scan_kernel<<<B_ * (DINNER/256), TPB, 0, stream>>>(
        dt_full, x_f32, xdbl, xz, A_log, D_param, y_bf);

    // 9) out = residual + y @ out_proj_w^T   (4096x1536x768)
    {
        int tiles = (NROWS/64) * (DMODEL/64);
        wmma_gemm_bf16_kernel<4,4><<<(tiles + 7)/8, TPB, 0, stream>>>(
            y_bf, outw_bf, hidden, out, NROWS, DMODEL, DINNER);
    }
}
